// QuantumStateEncoder_41764261986824
// MI455X (gfx1250) — matrix operand order repaired and checked by reading, not compiled, not run on hardware
//
#include <hip/hip_runtime.h>
#include <hip/hip_bf16.h>
#include <math.h>

// ---- CDNA5 WMMA vector types ----
typedef __attribute__((ext_vector_type(16))) __bf16 v16bf;
typedef __attribute__((ext_vector_type(8)))  __bf16 v8bf;
typedef __attribute__((ext_vector_type(8)))  float  v8f;
typedef __attribute__((ext_vector_type(4)))  float  v4f;

#define NB    16384   // batch
#define DIN   4096    // input dim
#define NCOL  1024    // 2*Q
#define NQUB  256     // qubits

// Scheduling fence: forbid the scheduler from sinking loads across phase
// boundaries (keeps the software pipeline intact). No-op if unavailable.
#if defined(__has_builtin)
#if __has_builtin(__builtin_amdgcn_sched_barrier)
#define QSE_SCHED_FENCE() __builtin_amdgcn_sched_barrier(0)
#endif
#endif
#ifndef QSE_SCHED_FENCE
#define QSE_SCHED_FENCE()
#endif

// -------------------------------------------------------------------------
// Kernel 0: build per-qubit 2x2 complex rotation R = Rz @ Ry @ Rx
// -------------------------------------------------------------------------
__global__ void qse_make_rot(const float* __restrict__ rp, float* __restrict__ Rm) {
    int n = blockIdx.x * blockDim.x + threadIdx.x;
    if (n >= NQUB) return;
    float rx = rp[n * 3 + 0] * 0.5f;
    float ry = rp[n * 3 + 1] * 0.5f;
    float rz = rp[n * 3 + 2] * 0.5f;
    float cx = cosf(rx), sx = sinf(rx);
    float cy = cosf(ry), sy = sinf(ry);
    float cz = cosf(rz), sz = sinf(rz);
    float M00r =  cy * cx, M00i =  sy * sx;
    float M01r = -sy * cx, M01i = -cy * sx;
    float M10r =  sy * cx, M10i = -cy * sx;
    float M11r =  cy * cx, M11i = -sy * sx;
    Rm[n * 8 + 0] = cz * M00r + sz * M00i;
    Rm[n * 8 + 1] = cz * M00i - sz * M00r;
    Rm[n * 8 + 2] = cz * M01r + sz * M01i;
    Rm[n * 8 + 3] = cz * M01i - sz * M01r;
    Rm[n * 8 + 4] = cz * M10r - sz * M10i;
    Rm[n * 8 + 5] = cz * M10i + sz * M10r;
    Rm[n * 8 + 6] = cz * M11r - sz * M11i;
    Rm[n * 8 + 7] = cz * M11i + sz * M11r;
}

// -------------------------------------------------------------------------
// Kernel 1: pack fp32 weight [K, 1024] into bf16 WMMA B-fragments.
// Per-lane v16bf: lanes 0-15 hold N=lane, K 0..7,16..23; lanes 16-31 hold
// N=lane-16, K 8..15,24..31 (ISA 16-bit B layout). One 32B load per fragment.
// -------------------------------------------------------------------------
__global__ void qse_pack_w(const float* __restrict__ W, __bf16* __restrict__ Wp, int K) {
    int tid  = blockIdx.x * blockDim.x + threadIdx.x;
    int fid  = tid >> 5;
    int lane = tid & 31;
    int nfrag = (K >> 5) * (NCOL >> 4);
    if (fid >= nfrag) return;
    int kt = fid / (NCOL >> 4);
    int nt = fid % (NCOL >> 4);
    int n     = nt * 16 + (lane & 15);
    int kbase = kt * 32 + (lane >> 4) * 8;
    __bf16* dst = Wp + ((size_t)fid * 32 + lane) * 16;
#pragma unroll
    for (int j = 0; j < 16; ++j)
        dst[j] = (__bf16)W[(size_t)(kbase + j + (j >> 3) * 8) * NCOL + n];
}

// ---- fragment load helpers ----
__device__ __forceinline__ v16bf qse_loadB(const v16bf* __restrict__ bp,
                                           int kt, int nt, int lane) {
    return bp[((size_t)kt * (NCOL / 16) + nt) * 32 + lane];
}

// Raw fp32 A loads (4x float4) -- issued in the load phase, converted later.
__device__ __forceinline__ void qse_loadAraw(const float* __restrict__ arow,
                                             int kt, v4f* q) {
    const float* p = arow + kt * 32;
    q[0] = *(const v4f*)(p);        // K 0-3   (or 8-11 for hi lanes)
    q[1] = *(const v4f*)(p + 4);    // K 4-7
    q[2] = *(const v4f*)(p + 16);   // K 16-19 (or 24-27)
    q[3] = *(const v4f*)(p + 20);   // K 20-23
}

__device__ __forceinline__ v16bf qse_cvtA(const v4f* q) {
    v16bf a;
#pragma unroll
    for (int i = 0; i < 4; ++i)
#pragma unroll
        for (int j = 0; j < 4; ++j)
            a[i * 4 + j] = (__bf16)q[i][j];
    return a;
}

// A-fragment from bf16 row-major (register-placement only, no VALU).
__device__ __forceinline__ v16bf qse_loadA_bf16(const __bf16* __restrict__ arow, int kt) {
    v8bf a0 = *(const v8bf*)(arow + kt * 32);
    v8bf a1 = *(const v8bf*)(arow + kt * 32 + 16);
    return __builtin_shufflevector(a0, a1, 0, 1, 2, 3, 4, 5, 6, 7,
                                   8, 9, 10, 11, 12, 13, 14, 15);
}

// -------------------------------------------------------------------------
// Kernel 2: H = tanh(X @ W1 + b1).
// Block = 16-row M tile x ALL 1024 columns (features read once from HBM).
// Wave w covers N = w*128..+127 (8 C fragments). K loop unrolled by 2 with
// sched_barrier-fenced load/compute phases: all loads of step k+1 issue
// before the WMMAs of step k, which only consume strictly-older loads, so
// waits are loadcnt<=N (pipelined), never a full drain.
// -------------------------------------------------------------------------
__global__ __launch_bounds__(256) void qse_gemm1(
    const float*  __restrict__ A,
    const __bf16* __restrict__ Bp,
    const float*  __restrict__ bias,
    __bf16*       __restrict__ H) {
    const int lane  = threadIdx.x & 31;
    const int wave  = threadIdx.x >> 5;
    const int m0    = blockIdx.x << 4;
    const int ntb   = wave * 8;
    const int row   = lane & 15;
    const int khalf = lane >> 4;
    const int STEPS = DIN / 32;          // 128 (even)

    const float* arow = A + (size_t)(m0 + row) * DIN + khalf * 8;
    const v16bf* bp   = (const v16bf*)Bp;

    v8f   c[8] = {};
    v16bf ba[8], bb[8];
    v4f   ar0[4], ar1[4];

    qse_loadAraw(arow, 0, ar0);
#pragma unroll
    for (int f = 0; f < 8; ++f) ba[f] = qse_loadB(bp, 0, ntb + f, lane);

    for (int kt = 0; kt < STEPS; kt += 2) {
        const int k1 = kt + 1;                                   // < STEPS
        const int k2 = (kt + 2 < STEPS) ? kt + 2 : STEPS - 1;
        const int kp = (kt + 8 < STEPS) ? kt + 8 : STEPS - 1;
        __builtin_prefetch(arow + kp * 32, 0, 3);   // global_prefetch_b8

        // ---- L phase: issue step k+1 operand loads ----
#pragma unroll
        for (int f = 0; f < 8; ++f) bb[f] = qse_loadB(bp, k1, ntb + f, lane);
        qse_loadAraw(arow, k1, ar1);
        QSE_SCHED_FENCE();
        // ---- C phase: consume step k (all operands strictly older) ----
        {
            v16bf a = qse_cvtA(ar0);
#pragma unroll
            for (int f = 0; f < 8; ++f)
                c[f] = __builtin_amdgcn_wmma_f32_16x16x32_bf16(false, a, false, ba[f],
                                                               (short)0, c[f], false, false);
        }
        QSE_SCHED_FENCE();
        // ---- L phase: issue step k+2 operand loads ----
#pragma unroll
        for (int f = 0; f < 8; ++f) ba[f] = qse_loadB(bp, k2, ntb + f, lane);
        qse_loadAraw(arow, k2, ar0);
        QSE_SCHED_FENCE();
        // ---- C phase: consume step k+1 ----
        {
            v16bf a = qse_cvtA(ar1);
#pragma unroll
            for (int f = 0; f < 8; ++f)
                c[f] = __builtin_amdgcn_wmma_f32_16x16x32_bf16(false, a, false, bb[f],
                                                               (short)0, c[f], false, false);
        }
        QSE_SCHED_FENCE();
    }

    // C layout: VGPR r, lanes 0-15 -> M=r, lanes 16-31 -> M=8+r; N = lane&15.
    const int ncol = wave * 128 + row;
#pragma unroll
    for (int f = 0; f < 8; ++f) {
        float bv = bias[ncol + f * 16];
#pragma unroll
        for (int r = 0; r < 8; ++r) {
            float v = tanhf(c[f][r] + bv);
            H[(size_t)(m0 + khalf * 8 + r) * NCOL + ncol + f * 16] = (__bf16)v;
        }
    }
}

// -------------------------------------------------------------------------
// Kernel 3 (fused): amp = H @ W2 + b2, per-row L2 normalization, per-qubit
// 2x2 complex rotation, writing d_out directly. Same fenced pipeline.
// -------------------------------------------------------------------------
__global__ __launch_bounds__(256) void qse_gemm2_fused(
    const __bf16* __restrict__ Hm,
    const __bf16* __restrict__ Bp,
    const float*  __restrict__ bias,
    const float*  __restrict__ Rm,
    float*        __restrict__ out) {
    __shared__ float tile[16 * NCOL];   // 64 KB amp strip
    __shared__ float rss[16];
    __shared__ float scl[16];

    const int tid   = threadIdx.x;
    const int lane  = tid & 31;
    const int wave  = tid >> 5;
    const int m0    = blockIdx.x << 4;
    const int ntb   = wave * 8;
    const int row   = lane & 15;
    const int khalf = lane >> 4;
    const int STEPS = NCOL / 32;        // 32 (even)

    if (tid < 16) rss[tid] = 0.0f;

    const __bf16* arow = Hm + (size_t)(m0 + row) * NCOL + khalf * 8;
    const v16bf*  bp   = (const v16bf*)Bp;

    v8f   c[8] = {};
    v16bf ba[8], bb[8];
    v16bf a0 = qse_loadA_bf16(arow, 0);
#pragma unroll
    for (int f = 0; f < 8; ++f) ba[f] = qse_loadB(bp, 0, ntb + f, lane);

    for (int kt = 0; kt < STEPS; kt += 2) {
        const int k1 = kt + 1;
        const int k2 = (kt + 2 < STEPS) ? kt + 2 : STEPS - 1;
        const int kp = (kt + 4 < STEPS) ? kt + 4 : STEPS - 1;
        __builtin_prefetch(arow + kp * 32, 0, 3);

        // L phase (k+1)
        v16bf a1 = qse_loadA_bf16(arow, k1);
#pragma unroll
        for (int f = 0; f < 8; ++f) bb[f] = qse_loadB(bp, k1, ntb + f, lane);
        QSE_SCHED_FENCE();
        // C phase (k)
#pragma unroll
        for (int f = 0; f < 8; ++f)
            c[f] = __builtin_amdgcn_wmma_f32_16x16x32_bf16(false, a0, false, ba[f],
                                                           (short)0, c[f], false, false);
        QSE_SCHED_FENCE();
        // L phase (k+2)
        a0 = qse_loadA_bf16(arow, k2);
#pragma unroll
        for (int f = 0; f < 8; ++f) ba[f] = qse_loadB(bp, k2, ntb + f, lane);
        QSE_SCHED_FENCE();
        // C phase (k+1)
#pragma unroll
        for (int f = 0; f < 8; ++f)
            c[f] = __builtin_amdgcn_wmma_f32_16x16x32_bf16(false, a1, false, bb[f],
                                                           (short)0, c[f], false, false);
        QSE_SCHED_FENCE();
    }

    // bias + stage amp into LDS + per-row sum-of-squares from registers
    const int ncol = wave * 128 + row;
    float bv[8];
#pragma unroll
    for (int f = 0; f < 8; ++f) bv[f] = bias[ncol + f * 16];

    __syncthreads();   // rss initialized, all waves present
#pragma unroll
    for (int r = 0; r < 8; ++r) {
        const int m = khalf * 8 + r;
        float part = 0.0f;
#pragma unroll
        for (int f = 0; f < 8; ++f) {
            float v = c[f][r] + bv[f];
            tile[m * NCOL + ncol + f * 16] = v;
            part += v * v;
        }
        atomicAdd(&rss[m], part);   // ds_add_f32
    }
    __syncthreads();
    if (tid < 16) scl[tid] = 1.0f / (sqrtf(rss[tid]) + 1e-8f);
    __syncthreads();

    // rotation: thread t = qubit t; float4 = (c0.re,c0.im,c1.re,c1.im)
    const float* r = Rm + tid * 8;
    const float r0 = r[0], r1 = r[1], r2 = r[2], r3 = r[3];
    const float r4 = r[4], r5 = r[5], r6 = r[6], r7 = r[7];
#pragma unroll 4
    for (int m = 0; m < 16; ++m) {
        const float  s = scl[m];
        const float4 v = ((const float4*)(tile + m * NCOL))[tid];
        const float c0r = v.x * s, c0i = v.y * s;
        const float c1r = v.z * s, c1i = v.w * s;
        float4 o;
        o.x = r0 * c0r - r1 * c0i + r2 * c1r - r3 * c1i;
        o.y = r0 * c0i + r1 * c0r + r2 * c1i + r3 * c1r;
        o.z = r4 * c0r - r5 * c0i + r6 * c1r - r7 * c1i;
        o.w = r4 * c0i + r5 * c0r + r6 * c1i + r7 * c1r;
        ((float4*)(out + (size_t)(m0 + m) * NCOL))[tid] = o;
    }
}

// -------------------------------------------------------------------------
extern "C" void kernel_launch(void* const* d_in, const int* in_sizes, int n_in,
                              void* d_out, int out_size, void* d_ws, size_t ws_size,
                              hipStream_t stream) {
    (void)in_sizes; (void)n_in; (void)out_size; (void)ws_size;
    const float* features = (const float*)d_in[0];  // [16384, 4096]
    const float* w1       = (const float*)d_in[1];  // [4096, 1024]
    const float* b1       = (const float*)d_in[2];  // [1024]
    const float* w2       = (const float*)d_in[3];  // [1024, 1024]
    const float* b2       = (const float*)d_in[4];  // [1024]
    const float* rot      = (const float*)d_in[5];  // [256, 3]
    float* out            = (float*)d_out;          // [16384, 1024]

    // workspace carve-up (all slices 256B aligned): ~42 MB total
    char* ws = (char*)d_ws;
    __bf16* w1p = (__bf16*)ws;  ws += (size_t)DIN  * NCOL * sizeof(__bf16);
    __bf16* w2p = (__bf16*)ws;  ws += (size_t)NCOL * NCOL * sizeof(__bf16);
    __bf16* H   = (__bf16*)ws;  ws += (size_t)NB   * NCOL * sizeof(__bf16);
    float*  Rm  = (float*)ws;   ws += (size_t)NQUB * 8    * sizeof(float);

    qse_make_rot<<<1, 256, 0, stream>>>(rot, Rm);
    qse_pack_w<<<((DIN  / 32) * (NCOL / 16) * 32) / 256, 256, 0, stream>>>(w1, w1p, DIN);
    qse_pack_w<<<((NCOL / 32) * (NCOL / 16) * 32) / 256, 256, 0, stream>>>(w2, w2p, NCOL);
    // one block per 16-row M tile, full 1024 columns per block
    qse_gemm1<<<NB / 16, 256, 0, stream>>>(features, w1p, b1, H);
    qse_gemm2_fused<<<NB / 16, 256, 0, stream>>>(H, w2p, b2, Rm, out);
}
